// ScaledDotProductAttention_29300266893596
// MI455X (gfx1250) — compile-verified, matrix operand order checked by reading, not run on hardware
//
#include <hip/hip_runtime.h>

// Scaled dot-product attention for MI455X (gfx1250), wave32 + WMMA f16.
// d_out = [ output B*S*D | attn_weights B*S*S ] (fp32).
// d_ws usage: 12 MB -> Qh (4MB) | Kh (4MB) | Vt (4MB, f16 transposed [B][D][S]).

typedef __attribute__((ext_vector_type(16))) _Float16 v16h;
typedef __attribute__((ext_vector_type(8)))  _Float16 v8h;
typedef __attribute__((ext_vector_type(8)))  float    v8f;

constexpr int Bc = 8;
constexpr int Sc = 2048;
constexpr int Dc = 128;
constexpr int QTILE = 64;                 // query rows per block (4 waves x 16)
constexpr float SCALE_LOG2E = 0.08838834764831845f * 1.4426950408889634f;

union Frag { v16h v; v8h h[2]; };

// ---- precision conversion: Q,K (fully coalesced) ---------------------------
__global__ __launch_bounds__(256) void cvt_qk_kernel(
    const float* __restrict__ q, const float* __restrict__ k,
    _Float16* __restrict__ qh, _Float16* __restrict__ kh, int n) {
  int i = blockIdx.x * blockDim.x + threadIdx.x;
  if (i < n) { qh[i] = (_Float16)q[i]; kh[i] = (_Float16)k[i]; }
}

// ---- V transpose via LDS tiles: Vt[b][d][n] = (f16)V[b][n][d] --------------
// 32x32 tile per block; coalesced 128B reads along d, coalesced 64B writes
// along n; padded LDS row (33) to avoid bank conflicts on transposed reads.
__global__ __launch_bounds__(256) void cvt_vt_kernel(
    const float* __restrict__ v, _Float16* __restrict__ vt) {
  __shared__ _Float16 tile[32][33];

  const int tilesD = Dc / 32;                         // 4
  const int tilesS = Sc / 32;                         // 64
  int t  = blockIdx.x;
  const int dt = t % tilesD;  t /= tilesD;
  const int nt = t % tilesS;  t /= tilesS;
  const int b  = t;                                   // grid = B * tilesS * tilesD
  const int d0 = dt * 32;
  const int n0 = nt * 32;

  const int dx = threadIdx.x & 31;                    // 0..31
  const int y8 = threadIdx.x >> 5;                    // 0..7

  // load: rows n0+y, columns d0+dx (coalesced along d)
#pragma unroll
  for (int p = 0; p < 4; ++p) {
    const int n = y8 + p * 8;
    tile[n][dx] = (_Float16)v[((size_t)b * Sc + (n0 + n)) * Dc + (d0 + dx)];
  }
  __syncthreads();

  // store: rows d0+y, columns n0+dx (coalesced along n)
#pragma unroll
  for (int p = 0; p < 4; ++p) {
    const int d = y8 + p * 8;
    vt[((size_t)b * Dc + (d0 + d)) * Sc + (n0 + dx)] = tile[dx][d];
  }
}

// ---- main attention kernel --------------------------------------------------
__global__ __launch_bounds__(128) void sdpa_wmma_kernel(
    const _Float16* __restrict__ Qh,   // [B][S][D] f16
    const _Float16* __restrict__ Kh,   // [B][S][D] f16
    const _Float16* __restrict__ Vt,   // [B][D][S] f16 (transposed)
    float* __restrict__ O,             // [B][S][D] f32
    float* __restrict__ W)             // [B][S][S] f32
{
  __shared__ _Float16 pstage[4][16][32];   // per-wave P tile (C-layout -> A-layout bounce)

  const int wave = threadIdx.x >> 5;
  const int lane = threadIdx.x & 31;
  const int ln   = lane & 15;              // row (A) / column (B) within tile
  const int hi   = lane >> 4;              // selects +8 k-chunk per ISA layout
  const int b    = blockIdx.x / (Sc / QTILE);
  const int q0   = (blockIdx.x % (Sc / QTILE)) * QTILE + wave * 16;

  // --- Q A-fragments: 4 chunks of k=32, resident in VGPRs for whole kernel ---
  Frag qa[4];
  {
    const _Float16* qrow = Qh + ((size_t)b * Sc + (q0 + ln)) * Dc;
#pragma unroll
    for (int kc = 0; kc < 4; ++kc) {
      const int kb = kc * 32 + hi * 8;
      qa[kc].h[0] = *(const v8h*)(qrow + kb);
      qa[kc].h[1] = *(const v8h*)(qrow + kb + 16);
    }
  }

  // scores tile for 16 keys starting at nt*16, D=128 contraction = 4 chained WMMAs
  auto score_tile = [&](int nt) -> v8f {
    v8f acc = {};
    const _Float16* krow = Kh + ((size_t)b * Sc + (nt * 16 + ln)) * Dc;
#pragma unroll
    for (int kc = 0; kc < 4; ++kc) {
      Frag kf;
      const int kb = kc * 32 + hi * 8;
      kf.h[0] = *(const v8h*)(krow + kb);
      kf.h[1] = *(const v8h*)(krow + kb + 16);
      acc = __builtin_amdgcn_wmma_f32_16x16x32_f16(
          false, qa[kc].v, false, kf.v, (short)0, acc, false, false);
    }
    return acc;
  };

  // ---------------- pass 1: row max (per-lane, then 16-lane xor reduce) ------
  float rmax[8];
#pragma unroll
  for (int r = 0; r < 8; ++r) rmax[r] = -3.0e30f;
  for (int nt = 0; nt < Sc / 16; ++nt) {
    v8f s = score_tile(nt);
#pragma unroll
    for (int r = 0; r < 8; ++r) rmax[r] = fmaxf(rmax[r], s[r] * SCALE_LOG2E);
  }
#pragma unroll
  for (int m = 1; m < 16; m <<= 1)
#pragma unroll
    for (int r = 0; r < 8; ++r) rmax[r] = fmaxf(rmax[r], __shfl_xor(rmax[r], m, 32));

  // ---------------- pass 2: exp-sum + O = P*V (unnormalized) -----------------
  float lsum[8];
  v8f oacc[8];
#pragma unroll
  for (int r = 0; r < 8; ++r) lsum[r] = 0.0f;
#pragma unroll
  for (int dt = 0; dt < 8; ++dt) oacc[dt] = (v8f){};

  for (int np = 0; np < Sc / 32; ++np) {   // 32 keys per PV step
#pragma unroll
    for (int half = 0; half < 2; ++half) {
      v8f s = score_tile(np * 2 + half);
#pragma unroll
      for (int r = 0; r < 8; ++r) {
        float p = exp2f(s[r] * SCALE_LOG2E - rmax[r]);
        lsum[r] += p;
        pstage[wave][r + 8 * hi][half * 16 + ln] = (_Float16)p;   // C-layout scatter
      }
    }
    asm volatile("s_wait_dscnt 0" ::: "memory");   // cross-lane LDS RAW fence (wave-local)

    Frag pa;                                        // P in A-layout: row=ln, k chunks
    {
      const _Float16* prow = &pstage[wave][ln][0];
      pa.h[0] = *(const v8h*)(prow + hi * 8);
      pa.h[1] = *(const v8h*)(prow + 16 + hi * 8);
    }
    const _Float16* vbase = Vt + ((size_t)b * Dc + ln) * Sc + np * 32;
#pragma unroll
    for (int dt = 0; dt < 8; ++dt) {
      Frag vf;                                      // B = V[32 x 16]: col=d, rows=n
      const _Float16* vrow = vbase + (size_t)dt * 16 * Sc;
      vf.h[0] = *(const v8h*)(vrow + hi * 8);
      vf.h[1] = *(const v8h*)(vrow + 16 + hi * 8);
      oacc[dt] = __builtin_amdgcn_wmma_f32_16x16x32_f16(
          false, pa.v, false, vf.v, (short)0, oacc[dt], false, false);
    }
  }

#pragma unroll
  for (int m = 1; m < 16; m <<= 1)
#pragma unroll
    for (int r = 0; r < 8; ++r) lsum[r] += __shfl_xor(lsum[r], m, 32);
  float inv[8];
#pragma unroll
  for (int r = 0; r < 8; ++r) inv[r] = 1.0f / lsum[r];

  // ---------------- pass 3: stream normalized attention weights -------------
  // Non-temporal: W is 134 MB write-once, keep 12 MB Qh/Kh/Vt resident in L2.
  for (int nt = 0; nt < Sc / 16; ++nt) {
    v8f s = score_tile(nt);
#pragma unroll
    for (int r = 0; r < 8; ++r) {
      const int m = r + 8 * hi;
      float p = exp2f(s[r] * SCALE_LOG2E - rmax[r]) * inv[r];
      __builtin_nontemporal_store(p, &W[((size_t)b * Sc + q0 + m) * Sc + nt * 16 + ln]);
    }
  }

  // ---------------- normalized output ----------------------------------------
#pragma unroll
  for (int dt = 0; dt < 8; ++dt)
#pragma unroll
    for (int r = 0; r < 8; ++r) {
      const int m = r + 8 * hi;
      __builtin_nontemporal_store(oacc[dt][r] * inv[r],
          &O[((size_t)b * Sc + q0 + m) * Dc + dt * 16 + ln]);
    }
}

// ---- host launch ------------------------------------------------------------
extern "C" void kernel_launch(void* const* d_in, const int* in_sizes, int n_in,
                              void* d_out, int out_size, void* d_ws, size_t ws_size,
                              hipStream_t stream) {
  const float* q = (const float*)d_in[0];
  const float* k = (const float*)d_in[1];
  const float* v = (const float*)d_in[2];

  float* out  = (float*)d_out;                       // [B,S,D]
  float* attn = out + (size_t)Bc * Sc * Dc;          // [B,S,S]

  const size_t elems = (size_t)Bc * Sc * Dc;         // 2,097,152
  _Float16* qh = (_Float16*)d_ws;                    // needs 3*elems*2 = 12 MB of ws
  _Float16* kh = qh + elems;
  _Float16* vt = kh + elems;

  const int n = (int)elems;
  const int cb = 256;
  cvt_qk_kernel<<<(n + cb - 1) / cb, cb, 0, stream>>>(q, k, qh, kh, n);
  cvt_vt_kernel<<<Bc * (Sc / 32) * (Dc / 32), cb, 0, stream>>>(v, vt);

  sdpa_wmma_kernel<<<Bc * (Sc / QTILE), 128, 0, stream>>>(qh, kh, vt, out, attn);
}
